// MultiHeadSelfAttention_49160195670596
// MI455X (gfx1250) — matrix operand order repaired and checked by reading, not compiled, not run on hardware
//
#include <hip/hip_runtime.h>
#include <cstdint>
#include <cstddef>

typedef __bf16 bf16_t;
typedef __attribute__((ext_vector_type(16))) __bf16 v16bf;
typedef __attribute__((ext_vector_type(8)))  float  v8f;

constexpr int B = 16, L = 512, D = 128, H = 8;
constexpr int HB = H * B;   // 128 head-major rows
constexpr int E  = H * D;   // 1024
constexpr float SCALE = 0.08838834764831845f; // 1/sqrt(128)

constexpr int BROWS = 64;        // B-panel rows staged per block (N_blk)
constexpr int BPAD  = 8;         // bf16 pad -> 272B row stride (17 x 16B)
constexpr int BLD   = 128 + BPAD;

// ---------------- WMMA fragment loaders (bf16, 16x16x32) ----------------
// A (16x32): lanes 0-15 = rows, elems {0..7}=K0..7, {8..15}=K16..23;
//            lanes 16-31 same rows, K {8..15} and {24..31}.
__device__ __forceinline__ v16bf load_fragA(const bf16_t* __restrict__ base, int ld,
                                            int m0, int k0, int lane) {
  const int r = lane & 15, hi = lane >> 4;
  const bf16_t* p = base + (size_t)(m0 + r) * ld + k0 + 8 * hi;
  union { float4 f4[2]; v16bf v; } u;
  u.f4[0] = *reinterpret_cast<const float4*>(p);
  u.f4[1] = *reinterpret_cast<const float4*>(p + 16);
  return u.v;
}
// B (32x16) from the LDS-staged [N][K] panel: lanes 0-15 = cols, K0..7,16..23;
// lanes 16-31 same cols, K8..15,24..31 (matches the A fragment order).
__device__ __forceinline__ v16bf load_fragB_lds(const bf16_t* __restrict__ Bs,
                                                int n0s, int k0, int lane) {
  const int c = lane & 15, hi = lane >> 4;
  const bf16_t* p = Bs + (size_t)(n0s + c) * BLD + k0 + 8 * hi;
  union { float4 f4[2]; v16bf v; } u;
  u.f4[0] = *reinterpret_cast<const float4*>(p);
  u.f4[1] = *reinterpret_cast<const float4*>(p + 16);
  return u.v;
}

// Stage one 64x128 bf16 B-panel chunk Global->LDS with async-LDS loads.
// 256 threads: each lane moves 4x16B (ASYNCcnt-tracked, completes at s_wait_asynccnt).
__device__ __forceinline__ void stage_B(bf16_t* BsBuf, const bf16_t* __restrict__ Bt,
                                        int ldb, int n_blk, int kc, int tid) {
  const int cc = (tid & 15) * 8;        // bf16 column within 128-wide chunk
#pragma unroll
  for (int pass = 0; pass < 4; ++pass) {
    const int rr = pass * 16 + (tid >> 4);
    const bf16_t* g = Bt + (size_t)(n_blk + rr) * ldb + kc + cc;
    const uint32_t l = (uint32_t)(uintptr_t)(BsBuf + (size_t)rr * BLD + cc);
    asm volatile("global_load_async_to_lds_b128 %0, %1, off"
                 :: "v"(l), "v"(g) : "memory");
  }
}
__device__ __forceinline__ void wait_async_and_sync() {
  asm volatile("s_wait_asynccnt 0x0" ::: "memory");
  __syncthreads();
}

// Block-level GEMM: 8 waves as 4(M)x2(N) -> 128x64 macro-tile, 32x32 per wave.
// A[M][K] bf16 from global (L2), Bt[N][K] bf16 staged to LDS, double-buffered.
__device__ __forceinline__ void gemm_block(v8f c[2][2],
    const bf16_t* __restrict__ A, int lda, int m_blk,
    const bf16_t* __restrict__ Bt, int ldb, int n_blk,
    int K, bf16_t* Bs /*[2][BROWS*BLD]*/,
    int tid, int lane, int mw, int nw) {
  const int m0  = m_blk + mw * 32;
  const int n0s = nw * 32;                   // LDS-local column base
  const int nchunks = K >> 7;                // K / 128
  int cur = 0;

  stage_B(Bs, Bt, ldb, n_blk, 0, tid);
  wait_async_and_sync();

  for (int ch = 0; ch < nchunks; ++ch) {
    const int kc = ch << 7;
    if (ch + 1 < nchunks)
      stage_B(Bs + (cur ^ 1) * (BROWS * BLD), Bt, ldb, n_blk, kc + 128, tid);

    const bf16_t* bs = Bs + cur * (BROWS * BLD);
#pragma unroll
    for (int k = 0; k < 128; k += 32) {
      v16bf a0 = load_fragA(A, lda, m0,      kc + k, lane);
      v16bf a1 = load_fragA(A, lda, m0 + 16, kc + k, lane);
      v16bf b0 = load_fragB_lds(bs, n0s,      k, lane);
      v16bf b1 = load_fragB_lds(bs, n0s + 16, k, lane);
      c[0][0] = __builtin_amdgcn_wmma_f32_16x16x32_bf16(false, a0, false, b0, (short)0, c[0][0], false, false);
      c[0][1] = __builtin_amdgcn_wmma_f32_16x16x32_bf16(false, a0, false, b1, (short)0, c[0][1], false, false);
      c[1][0] = __builtin_amdgcn_wmma_f32_16x16x32_bf16(false, a1, false, b0, (short)0, c[1][0], false, false);
      c[1][1] = __builtin_amdgcn_wmma_f32_16x16x32_bf16(false, a1, false, b1, (short)0, c[1][1], false, false);
    }
    if (ch + 1 < nchunks) {
      wait_async_and_sync();                 // chunk ready + everyone done with old buf
      cur ^= 1;
    }
  }
}

// ---------------- helper kernels ----------------
__global__ void cvt_f32_bf16(const float* __restrict__ s, bf16_t* __restrict__ d, int n) {
  int i = blockIdx.x * blockDim.x + threadIdx.x;
  if (i < n) d[i] = (bf16_t)s[i];
}
__global__ void zero_f32_kernel(float* __restrict__ p, int n) {
  int i = blockIdx.x * blockDim.x + threadIdx.x;
  if (i < n) p[i] = 0.f;
}

// ---------------- K1: fused Q/K/V projections ----------------
__global__ void proj_kernel(const bf16_t* __restrict__ xb,
                            const bf16_t* __restrict__ wqb, const bf16_t* __restrict__ wkb,
                            const bf16_t* __restrict__ wvb,
                            const float* __restrict__ b_q, const float* __restrict__ b_k,
                            const float* __restrict__ b_v,
                            bf16_t* __restrict__ qb, bf16_t* __restrict__ kb,
                            bf16_t* __restrict__ vTb) {
  __shared__ bf16_t Bs[2][BROWS * BLD];
  const int tid = threadIdx.x, lane = tid & 31, wave = tid >> 5;
  const int mw = wave >> 1, nw = wave & 1;
  const int m_blk = (blockIdx.x >> 1) * 128, n_blk = (blockIdx.x & 1) * 64;
  const int nidx = blockIdx.y;                   // h*B + b
  const int h = nidx / B, b = nidx % B;
  const int z = blockIdx.z;                      // 0=q 1=k 2=v

  const bf16_t* W    = (z == 0) ? wqb : (z == 1) ? wkb : wvb;
  const float*  bias = (z == 0) ? b_q : (z == 1) ? b_k : b_v;
  const bf16_t* A  = xb + (size_t)b * L * D;     // [512][128]
  const bf16_t* Bt = W + (size_t)h * D * D;      // head's 128 rows of [E][D]

  v8f c[2][2] = {};
  gemm_block(c, A, D, m_blk, Bt, D, n_blk, D, &Bs[0][0], tid, lane, mw, nw);

  const int ln = lane & 15, hi = lane >> 4;
  for (int j = 0; j < 2; ++j) {
    const int col = n_blk + nw * 32 + 16 * j + ln;          // d' within head
    const float bv = bias[h * D + col];
    for (int i = 0; i < 2; ++i)
      for (int r = 0; r < 8; ++r) {
        const int m = m_blk + mw * 32 + 16 * i + 8 * hi + r;  // l
        const float y = c[i][j][r] + bv;
        if (z == 0)      qb [(size_t)nidx * L * D + (size_t)m * D + col] = (bf16_t)y;
        else if (z == 1) kb [(size_t)nidx * L * D + (size_t)m * D + col] = (bf16_t)y;
        else             vTb[(size_t)nidx * D * L + (size_t)col * L + m] = (bf16_t)y;
      }
  }
}

// ---------------- K2: scores = qk^T, mask-before-scale, exp, column sums ----
__global__ void scores_kernel(const bf16_t* __restrict__ qb, const bf16_t* __restrict__ kb,
                              const unsigned char* __restrict__ mask,
                              bf16_t* __restrict__ ebuf, float* __restrict__ colsum) {
  __shared__ bf16_t Bs[2][BROWS * BLD];
  const int tid = threadIdx.x, lane = tid & 31, wave = tid >> 5;
  const int mw = wave >> 1, nw = wave & 1;
  const int m_blk = (blockIdx.x >> 3) * 128, n_blk = (blockIdx.x & 7) * 64;
  const int nidx = blockIdx.y;
  const int b = nidx % B;                        // tiled mask: mask[n % B]

  const bf16_t* A  = qb + (size_t)nidx * L * D;
  const bf16_t* Bt = kb + (size_t)nidx * L * D;

  v8f c[2][2] = {};
  gemm_block(c, A, D, m_blk, Bt, D, n_blk, D, &Bs[0][0], tid, lane, mw, nw);

  const int ln = lane & 15, hi = lane >> 4;
  const unsigned char* mrow = mask + (size_t)b * L * L;
  bf16_t* erow = ebuf + (size_t)nidx * L * L;
  for (int j = 0; j < 2; ++j) {
    const int kcol = n_blk + nw * 32 + 16 * j + ln;
    float part = 0.f;
    for (int i = 0; i < 2; ++i)
      for (int r = 0; r < 8; ++r) {
        const int q = m_blk + mw * 32 + 16 * i + 8 * hi + r;
        float s = c[i][j][r];
        if (mrow[(size_t)q * L + kcol]) s = -1e9f;   // mask BEFORE scaling
        const float ev = __expf(s * SCALE);          // masked -> exp(-8.8e7)=0
        erow[(size_t)q * L + kcol] = (bf16_t)ev;
        part += ev;
      }
    atomicAdd(&colsum[nidx * L + kcol], part);       // softmax over QUERY axis
  }
}

// ---------------- K3: fold normalization into V (in place) ----------------
__global__ void scale_vT_kernel(bf16_t* __restrict__ vT, const float* __restrict__ colsum, int total) {
  int idx = blockIdx.x * blockDim.x + threadIdx.x;
  if (idx >= total) return;
  const int n = idx / (D * L);
  const int l = idx % L;                         // layout [n][d][l]
  const float cs = colsum[n * L + l];
  vT[idx] = (bf16_t)((float)vT[idx] / cs);
}

// ---------------- K4: att = E * (V/colsum), stored flat (H*B,L,D) -----------
__global__ void attv_kernel(const bf16_t* __restrict__ ebuf, const bf16_t* __restrict__ vTb,
                            bf16_t* __restrict__ attb) {
  __shared__ bf16_t Bs[2][BROWS * BLD];
  const int tid = threadIdx.x, lane = tid & 31, wave = tid >> 5;
  const int mw = wave >> 1, nw = wave & 1;
  const int m_blk = (blockIdx.x >> 1) * 128, n_blk = (blockIdx.x & 1) * 64;
  const int nidx = blockIdx.y;

  const bf16_t* A  = ebuf + (size_t)nidx * L * L;   // [512 q][512 k]
  const bf16_t* Bt = vTb  + (size_t)nidx * D * L;   // [128 d][512 k]

  v8f c[2][2] = {};
  gemm_block(c, A, L, m_blk, Bt, L, n_blk, L, &Bs[0][0], tid, lane, mw, nw);

  const int ln = lane & 15, hi = lane >> 4;
  for (int j = 0; j < 2; ++j) {
    const int col = n_blk + nw * 32 + 16 * j + ln;
    for (int i = 0; i < 2; ++i)
      for (int r = 0; r < 8; ++r) {
        const int m = m_blk + mw * 32 + 16 * i + 8 * hi + r;
        attb[(size_t)nidx * L * D + (size_t)m * D + col] = (bf16_t)c[i][j][r];
      }
  }
}

// ---------------- K5: output projection over the torch-view [8192][1024] ----
__global__ void out_kernel(const bf16_t* __restrict__ attb, const bf16_t* __restrict__ wob,
                           const float* __restrict__ b_o, float* __restrict__ out) {
  __shared__ bf16_t Bs[2][BROWS * BLD];
  const int tid = threadIdx.x, lane = tid & 31, wave = tid >> 5;
  const int mw = wave >> 1, nw = wave & 1;
  const int m_blk = (blockIdx.x >> 1) * 128, n_blk = (blockIdx.x & 1) * 64;

  v8f c[2][2] = {};
  gemm_block(c, attb, E, m_blk, wob, E, n_blk, E, &Bs[0][0], tid, lane, mw, nw);

  const int ln = lane & 15, hi = lane >> 4;
  for (int j = 0; j < 2; ++j) {
    const int col = n_blk + nw * 32 + 16 * j + ln;
    const float bo = b_o[col];
    for (int i = 0; i < 2; ++i)
      for (int r = 0; r < 8; ++r) {
        const int m = m_blk + mw * 32 + 16 * i + 8 * hi + r;
        out[(size_t)m * D + col] = c[i][j][r] + bo;
      }
  }
}

// ---------------- launch ----------------
extern "C" void kernel_launch(void* const* d_in, const int* in_sizes, int n_in,
                              void* d_out, int out_size, void* d_ws, size_t ws_size,
                              hipStream_t stream) {
  (void)in_sizes; (void)n_in; (void)out_size; (void)ws_size;
  const float* x   = (const float*)d_in[0];
  const float* W_q = (const float*)d_in[1];
  const float* b_q = (const float*)d_in[2];
  const float* W_k = (const float*)d_in[3];
  const float* b_k = (const float*)d_in[4];
  const float* W_v = (const float*)d_in[5];
  const float* b_v = (const float*)d_in[6];
  const float* W_o = (const float*)d_in[7];
  const float* b_o = (const float*)d_in[8];
  const unsigned char* mask = (const unsigned char*)d_in[9];
  float* out = (float*)d_out;

  char* base = (char*)d_ws;
  size_t off = 0;
  auto take = [&](size_t bytes) -> char* {
    char* q = base + off;
    off += (bytes + 255) & ~(size_t)255;
    return q;
  };
  bf16_t* xb   = (bf16_t*)take((size_t)B * L * D * 2);
  bf16_t* wqb  = (bf16_t*)take((size_t)E * D * 2);
  bf16_t* wkb  = (bf16_t*)take((size_t)E * D * 2);
  bf16_t* wvb  = (bf16_t*)take((size_t)E * D * 2);
  bf16_t* wob  = (bf16_t*)take((size_t)D * E * 2);
  bf16_t* qb   = (bf16_t*)take((size_t)HB * L * D * 2);
  bf16_t* kb   = (bf16_t*)take((size_t)HB * L * D * 2);
  bf16_t* vTb  = (bf16_t*)take((size_t)HB * D * L * 2);
  bf16_t* ebuf = (bf16_t*)take((size_t)HB * L * L * 2);
  float*  colsum = (float*)take((size_t)HB * L * 4);
  bf16_t* attb = (bf16_t*)take((size_t)HB * L * D * 2);

  cvt_f32_bf16<<<(B * L * D + 255) / 256, 256, 0, stream>>>(x,   xb,  B * L * D);
  cvt_f32_bf16<<<(E * D + 255) / 256,     256, 0, stream>>>(W_q, wqb, E * D);
  cvt_f32_bf16<<<(E * D + 255) / 256,     256, 0, stream>>>(W_k, wkb, E * D);
  cvt_f32_bf16<<<(E * D + 255) / 256,     256, 0, stream>>>(W_v, wvb, E * D);
  cvt_f32_bf16<<<(D * E + 255) / 256,     256, 0, stream>>>(W_o, wob, D * E);
  zero_f32_kernel<<<(HB * L + 255) / 256, 256, 0, stream>>>(colsum, HB * L);

  proj_kernel  <<<dim3(8, HB, 3), 256, 0, stream>>>(xb, wqb, wkb, wvb, b_q, b_k, b_v, qb, kb, vTb);
  scores_kernel<<<dim3(32, HB),   256, 0, stream>>>(qb, kb, mask, ebuf, colsum);
  scale_vT_kernel<<<(HB * D * L + 255) / 256, 256, 0, stream>>>(vTb, colsum, HB * D * L);
  attv_kernel  <<<dim3(8, HB),    256, 0, stream>>>(ebuf, vTb, attb);
  out_kernel   <<<128,            256, 0, stream>>>(attb, wob, b_o, out);
}
